// SingleHeadAttention_40029095198913
// MI455X (gfx1250) — matrix operand order repaired and checked by reading, not compiled, not run on hardware
//
#include <hip/hip_runtime.h>
#include <cmath>

typedef __attribute__((ext_vector_type(16))) __bf16 v16bf;
typedef __attribute__((ext_vector_type(8)))  __bf16 v8bf;
typedef __attribute__((ext_vector_type(8)))  float  v8f;

__device__ __forceinline__ unsigned short bf16bits(float f) {
  unsigned u = __builtin_bit_cast(unsigned, f);
  u += 0x7FFFu + ((u >> 16) & 1u);           // round-to-nearest-even
  return (unsigned short)(u >> 16);
}
__device__ __forceinline__ __bf16 f2bf(float f) {
  return __builtin_bit_cast(__bf16, bf16bits(f));
}

// ---------------------------------------------------------------------------
// fp32 -> bf16 bulk conversion (memory-bound, runs once per tensor).
// n4 = element count / 4.
// ---------------------------------------------------------------------------
__global__ __launch_bounds__(256) void cvt_bf16_kernel(
    const float* __restrict__ src, __bf16* __restrict__ dst, int n4) {
  const int i = blockIdx.x * 256 + threadIdx.x;
  if (i < n4) {
    const float4 f = ((const float4*)src)[i];
    ushort4 o;
    o.x = bf16bits(f.x); o.y = bf16bits(f.y);
    o.z = bf16bits(f.z); o.w = bf16bits(f.w);
    ((ushort4*)dst)[i] = o;
  }
}

// ---------------------------------------------------------------------------
// C[M,N] = A[M,K] * B[K,N] (+bias). A,B are bf16 row-major; C is bf16 or f32.
// lda=K, ldb=N, ldc=N. K multiple of 256, N multiple of 16.
// 256 threads = 8 waves; one 16x16 C tile per wave. B chunk (256x16) staged
// in LDS pre-swizzled into per-lane fragment layout (coalesced 32B global
// row reads, strided 16-bit LDS writes); each wave then reads its whole B
// fragment with a single 32B ds load. No conversions in the hot loop.
// ---------------------------------------------------------------------------
template <bool OUT_BF16>
__global__ __launch_bounds__(256) void gemm_bb_kernel(
    const __bf16* __restrict__ A, const __bf16* __restrict__ Bm,
    const float* __restrict__ bias, void* __restrict__ Cv,
    int M, int N, int K, long long sA, long long sB, long long sC) {
  const int b = blockIdx.z;
  A  += (size_t)b * sA;
  Bm += (size_t)b * sB;

  const int n0    = blockIdx.x * 16;
  const int m0    = blockIdx.y * 128;
  const int w     = threadIdx.x >> 5;
  const int lane  = threadIdx.x & 31;
  const int half  = lane >> 4;
  const int lm    = lane & 15;
  const int mtile = m0 + w * 16;
  const int t     = threadIdx.x;

  __shared__ __align__(32) unsigned short lb[8 * 32 * 16];  // 8 KB

  v8f acc = {};

  int arow = mtile + lm;
  if (arow >= M) arow = M - 1;                 // clamp; masked on store
  const __bf16* aptr = A + (size_t)arow * K;

  for (int kc = 0; kc < K; kc += 256) {
    // Stage B[kc+t, n0..n0+15]: one coalesced 32B row per thread.
    {
      const v8bf r0 = *(const v8bf*)(Bm + (size_t)(kc + t) * N + n0);
      const v8bf r1 = *(const v8bf*)(Bm + (size_t)(kc + t) * N + n0 + 8);
      const int ks = t >> 5, hf = (t >> 4) & 1, j = t & 15;
      unsigned short* dst = &lb[(ks * 32 + hf * 16) * 16 + j];
#pragma unroll
      for (int nn = 0; nn < 8; ++nn)
        dst[nn * 16] = __builtin_bit_cast(unsigned short, r0[nn]);
#pragma unroll
      for (int nn = 0; nn < 8; ++nn)
        dst[(8 + nn) * 16] = __builtin_bit_cast(unsigned short, r1[nn]);
    }
    __syncthreads();

#pragma unroll
    for (int ks = 0; ks < 8; ++ks) {
      const __bf16* ap = aptr + kc + ks * 32 + half * 16;
      const v8bf alo = *(const v8bf*)ap;         // 16B load
      const v8bf ahi = *(const v8bf*)(ap + 8);   // 16B load
      const v16bf af = __builtin_shufflevector(
          alo, ahi, 0, 1, 2, 3, 4, 5, 6, 7, 8, 9, 10, 11, 12, 13, 14, 15);
      const v16bf bf = *(const v16bf*)&lb[(ks * 32 + lane) * 16];  // 32B ds
      acc = __builtin_amdgcn_wmma_f32_16x16x32_bf16(
          false, af, false, bf, (short)0, acc, false, false);
    }
    __syncthreads();
  }

  const int   n  = n0 + lm;
  const float bv = bias ? bias[n] : 0.0f;
  const int mbase = mtile + half * 8;
  if (OUT_BF16) {
    __bf16* C = (__bf16*)Cv + (size_t)b * sC;
#pragma unroll
    for (int r = 0; r < 8; ++r) {
      const int m = mbase + r;
      if (m < M) C[(size_t)m * N + n] = f2bf(acc[r] + bv);
    }
  } else {
    float* C = (float*)Cv + (size_t)b * sC;
#pragma unroll
    for (int r = 0; r < 8; ++r) {
      const int m = mbase + r;
      if (m < M) C[(size_t)m * N + n] = acc[r] + bv;
    }
  }
}

// ---------------------------------------------------------------------------
// Fused scores + max-pool on bf16 qp/kp:
//   scores[b,q,k] = (qp[b,q,:] . kp[b,k,:]) / 16
//   pooled[b,p,k] = max over q in [p*10-2, p*10+7]  (SAME pad_before = 2)
// One workgroup: batch b, 160 query rows starting at qb*160-2, 8 k-tiles
// (one per wave, B fragments preloaded: contiguous 32B global loads).
// ---------------------------------------------------------------------------
__global__ __launch_bounds__(256) void scores_pool_kernel(
    const __bf16* __restrict__ qp, const __bf16* __restrict__ kp,
    float* __restrict__ P) {
  constexpr int S = 4096, D = 256, SP = 410;
  const int b    = blockIdx.z;
  const int qb   = blockIdx.y;                  // 0..25
  const int w    = threadIdx.x >> 5;
  const int lane = threadIdx.x & 31;
  const int half = lane >> 4;
  const int lm   = lane & 15;
  const int k0   = (blockIdx.x * 8 + w) * 16;   // k-tile base
  const int qbase = qb * 160 - 2;               // SAME pad shift

  const __bf16* Q  = qp + (size_t)b * S * D;
  const __bf16* Kp = kp + (size_t)b * S * D;

  __shared__ float scratch[8][16][16];
  __shared__ float pooled[8][16][16];
  for (int i = lane; i < 256; i += 32)
    pooled[w][i >> 4][i & 15] = -INFINITY;      // per-wave region, no barrier

  // Preload the 8 B fragments of this k-tile of Kp (contiguous bf16 rows).
  v16bf bfrag[8];
#pragma unroll
  for (int ks = 0; ks < 8; ++ks)
    bfrag[ks] = *(const v16bf*)(Kp + (size_t)(k0 + lm) * D + ks * 32 + half * 16);

  for (int t = 0; t < 10; ++t) {
    const int qrow = qbase + t * 16 + lm;
    int ar = qrow; if (ar < 0) ar = 0; if (ar >= S) ar = S - 1;
    const __bf16* arowp = Q + (size_t)ar * D;

    v8f acc = {};
#pragma unroll
    for (int ks = 0; ks < 8; ++ks) {
      const __bf16* ap = arowp + ks * 32 + half * 16;
      const v8bf alo = *(const v8bf*)ap;
      const v8bf ahi = *(const v8bf*)(ap + 8);
      const v16bf af = __builtin_shufflevector(
          alo, ahi, 0, 1, 2, 3, 4, 5, 6, 7, 8, 9, 10, 11, 12, 13, 14, 15);
      acc = __builtin_amdgcn_wmma_f32_16x16x32_bf16(
          false, af, false, bfrag[ks], (short)0, acc, false, false);
    }

    // Dump tile to LDS (scaled by 1/sqrt(256)); OOB query rows -> -inf.
#pragma unroll
    for (int r = 0; r < 8; ++r) {
      const int rl = r + half * 8;
      const int qg = qbase + t * 16 + rl;
      scratch[w][rl][lm] = (qg >= 0 && qg < S) ? acc[r] * 0.0625f : -INFINITY;
    }
    // Wave-internal pooling (lanes 0..15, one k column each; race-free).
    if (half == 0) {
      for (int r2 = 0; r2 < 16; ++r2) {
        const int win = (t * 16 + r2) / 10;     // 0..15
        const float v = scratch[w][r2][lm];
        if (v > pooled[w][win][lm]) pooled[w][win][lm] = v;
      }
    }
  }

#pragma unroll
  for (int r = 0; r < 8; ++r) {
    const int win = half * 8 + r;
    const int p = qb * 16 + win;
    if (p < SP)
      P[((size_t)b * SP + p) * S + k0 + lm] = pooled[w][win][lm];
  }
}

// ---------------------------------------------------------------------------
// Row softmax: read fp32 pooled scores, write bf16 attention matrix.
// rows = 4*410, cols = 4096; each thread keeps its 16 exps in registers.
// ---------------------------------------------------------------------------
__global__ __launch_bounds__(256) void softmax_kernel(
    const float* __restrict__ P, __bf16* __restrict__ Ab) {
  constexpr int C = 4096;
  const float* pr = P + (size_t)blockIdx.x * C;
  __bf16* ab = Ab + (size_t)blockIdx.x * C;
  __shared__ float red[256];
  const int t = threadIdx.x;

  float m = -INFINITY;
  for (int i = t; i < C; i += 256) m = fmaxf(m, pr[i]);
  red[t] = m; __syncthreads();
  for (int s = 128; s > 0; s >>= 1) {
    if (t < s) red[t] = fmaxf(red[t], red[t + s]);
    __syncthreads();
  }
  m = red[0]; __syncthreads();

  float ev[16];
  float sum = 0.0f;
#pragma unroll
  for (int idx = 0; idx < 16; ++idx) {
    const float e = __expf(pr[t + idx * 256] - m);
    ev[idx] = e;
    sum += e;
  }
  red[t] = sum; __syncthreads();
  for (int s = 128; s > 0; s >>= 1) {
    if (t < s) red[t] += red[t + s];
    __syncthreads();
  }
  const float inv = 1.0f / red[0];
#pragma unroll
  for (int idx = 0; idx < 16; ++idx)
    ab[t + idx * 256] = f2bf(ev[idx] * inv);
}

// ---------------------------------------------------------------------------
extern "C" void kernel_launch(void* const* d_in, const int* in_sizes, int n_in,
                              void* d_out, int out_size, void* d_ws, size_t ws_size,
                              hipStream_t stream) {
  const float* v  = (const float*)d_in[0];
  const float* k  = (const float*)d_in[1];
  const float* q  = (const float*)d_in[2];
  const float* wq = (const float*)d_in[3];
  const float* bq = (const float*)d_in[4];
  const float* wk = (const float*)d_in[5];
  const float* bk = (const float*)d_in[6];
  const float* wv = (const float*)d_in[7];
  const float* bv = (const float*)d_in[8];
  const float* wo = (const float*)d_in[9];
  const float* bo = (const float*)d_in[10];

  constexpr int B = 4, S = 4096, D = 256, SP = 410;
  constexpr size_t BSD = (size_t)B * S * D;   // 4,194,304
  constexpr size_t DD  = (size_t)D * D;       // 65,536
  constexpr size_t PSZ = (size_t)B * SP * S;  // 6,717,440
  constexpr size_t OSZ = (size_t)B * SP * D;  // 419,840

  char* wsb = (char*)d_ws;
  auto alloc = [&](size_t bytes) {
    char* p = wsb;
    wsb += (bytes + 255) & ~(size_t)255;
    return p;
  };
  __bf16* qc  = (__bf16*)alloc(BSD * 2);
  __bf16* kc  = (__bf16*)alloc(BSD * 2);
  __bf16* vc  = (__bf16*)alloc(BSD * 2);
  __bf16* wqc = (__bf16*)alloc(DD * 2);
  __bf16* wkc = (__bf16*)alloc(DD * 2);
  __bf16* wvc = (__bf16*)alloc(DD * 2);
  __bf16* woc = (__bf16*)alloc(DD * 2);
  __bf16* qp  = (__bf16*)alloc(BSD * 2);
  __bf16* kp  = (__bf16*)alloc(BSD * 2);
  __bf16* vp  = (__bf16*)alloc(BSD * 2);
  float*  P   = (float*)alloc(PSZ * 4);
  __bf16* Ab  = (__bf16*)alloc(PSZ * 2);
  __bf16* Ob  = (__bf16*)alloc(OSZ * 2);

  dim3 blk(256);

  // 0) one-time fp32 -> bf16 conversions
  cvt_bf16_kernel<<<dim3(BSD / 4 / 256), blk, 0, stream>>>(q, qc, BSD / 4);
  cvt_bf16_kernel<<<dim3(BSD / 4 / 256), blk, 0, stream>>>(k, kc, BSD / 4);
  cvt_bf16_kernel<<<dim3(BSD / 4 / 256), blk, 0, stream>>>(v, vc, BSD / 4);
  cvt_bf16_kernel<<<dim3(DD / 4 / 256), blk, 0, stream>>>(wq, wqc, DD / 4);
  cvt_bf16_kernel<<<dim3(DD / 4 / 256), blk, 0, stream>>>(wk, wkc, DD / 4);
  cvt_bf16_kernel<<<dim3(DD / 4 / 256), blk, 0, stream>>>(wv, wvc, DD / 4);
  cvt_bf16_kernel<<<dim3(DD / 4 / 256), blk, 0, stream>>>(wo, woc, DD / 4);

  // 1) projections -> bf16 qp/kp/vp : [16384,256] x [256,256] (+bias)
  gemm_bb_kernel<true><<<dim3(D / 16, (B * S) / 128, 1), blk, 0, stream>>>(
      qc, wqc, bq, qp, B * S, D, D, 0, 0, 0);
  gemm_bb_kernel<true><<<dim3(D / 16, (B * S) / 128, 1), blk, 0, stream>>>(
      kc, wkc, bk, kp, B * S, D, D, 0, 0, 0);
  gemm_bb_kernel<true><<<dim3(D / 16, (B * S) / 128, 1), blk, 0, stream>>>(
      vc, wvc, bv, vp, B * S, D, D, 0, 0, 0);

  // 2) fused scores (qp . kp^T / 16) + max-pool over query axis -> P (fp32)
  scores_pool_kernel<<<dim3(S / (16 * 8), 26, B), blk, 0, stream>>>(qp, kp, P);

  // 3) softmax rows -> bf16 attention
  softmax_kernel<<<dim3(B * SP), blk, 0, stream>>>(P, Ab);

  // 4) attn @ vp : per-batch [410,4096] x [4096,256] -> bf16
  gemm_bb_kernel<true><<<dim3(D / 16, (SP + 127) / 128, B), blk, 0, stream>>>(
      Ab, vp, nullptr, Ob, SP, D, S,
      (long long)SP * S, (long long)S * D, (long long)SP * D);

  // 5) final projection: [1640,256] x [256,256] + bo -> d_out (fp32)
  gemm_bb_kernel<false><<<dim3(D / 16, (B * SP + 127) / 128, 1), blk, 0, stream>>>(
      Ob, woc, bo, (float*)d_out, B * SP, D, D, 0, 0, 0);
}
